// GIN_52819507806389
// MI455X (gfx1250) — matrix operand order repaired and checked by reading, not compiled, not run on hardware
//
#include <hip/hip_runtime.h>
#include <hip/hip_bf16.h>

#define DIM 128
#define N_NODES_C 50000
#define N_EDGES_C 800000
#define N_GRAPHS_C 64
#define N_CLASSES_C 10

typedef __attribute__((ext_vector_type(16))) __bf16 v16bf;
typedef __attribute__((ext_vector_type(8)))  __bf16 v8bf;
typedef __attribute__((ext_vector_type(4)))  __bf16 v4bf;
typedef __attribute__((ext_vector_type(8)))  float  v8f;

// XOR bank swizzle: flips column bits 4..6 by row&7. Keeps any 16-element
// aligned chunk contiguous, spreads rows over 8 LDS bank groups.
__device__ __forceinline__ int swz(int row, int col) {
    return row * DIM + (col ^ ((row & 7) << 4));
}

// ---------------------------------------------------------------------------
// zero fill
// ---------------------------------------------------------------------------
__global__ void gin_zero_f32(float* __restrict__ p, int n) {
    int i = blockIdx.x * blockDim.x + threadIdx.x;
    if (i < n) p[i] = 0.0f;
}

// ---------------------------------------------------------------------------
// scatter: agg[dst] += x[src]   (32 lanes per edge, float4 per lane,
// edge indices forced wave-uniform -> SMEM loads)
// ---------------------------------------------------------------------------
__global__ __launch_bounds__(256) void gin_scatter(
    const float* __restrict__ x,
    const long long* __restrict__ src,
    const long long* __restrict__ dst,
    float* __restrict__ agg, int nEdges)
{
    int e = __builtin_amdgcn_readfirstlane(blockIdx.x * 8 + (threadIdx.x >> 5));
    int j = threadIdx.x & 31;
    if (e >= nEdges) return;
    int s = (int)src[e];   // wave-uniform -> scalar load
    int d = (int)dst[e];
    const float4 v = *(const float4*)&x[(long long)s * DIM + j * 4];
    float* base = &agg[(long long)d * DIM + j * 4];
    unsafeAtomicAdd(base + 0, v.x);
    unsafeAtomicAdd(base + 1, v.y);
    unsafeAtomicAdd(base + 2, v.z);
    unsafeAtomicAdd(base + 3, v.w);
}

// ---------------------------------------------------------------------------
// fused GIN MLP: out = relu(relu((x+agg)@Wa + ba)@Wb + bb)
// one 128-row tile per workgroup, bf16 WMMA, LDS-resident intermediates
// ---------------------------------------------------------------------------
__global__ __launch_bounds__(256) void gin_mlp(
    const float* __restrict__ xin, const float* __restrict__ agg,
    const float* __restrict__ Wa,  const float* __restrict__ ba,
    const float* __restrict__ Wb,  const float* __restrict__ bb,
    float* __restrict__ out, int nRows)
{
    __shared__ __align__(32) __bf16 sAct[DIM * DIM];   // 32 KB activations
    __shared__ __align__(32) __bf16 sW[DIM * DIM];     // 32 KB W^T

    const int tid = threadIdx.x;
    const int r0  = blockIdx.x * DIM;

    // ---- stage activations: (x + agg) -> bf16, float4 loads, b64 stores ----
    for (int idx4 = tid; idx4 < DIM * DIM / 4; idx4 += 256) {
        int r  = idx4 >> 5;
        int c4 = (idx4 & 31) * 4;
        int row = r0 + r;
        v4bf o;
        if (row < nRows) {
            long long off = (long long)row * DIM + c4;
            const float4 xv = *(const float4*)&xin[off];
            const float4 av = *(const float4*)&agg[off];
            o[0] = (__bf16)(xv.x + av.x);
            o[1] = (__bf16)(xv.y + av.y);
            o[2] = (__bf16)(xv.z + av.z);
            o[3] = (__bf16)(xv.w + av.w);
        } else {
            o[0] = o[1] = o[2] = o[3] = (__bf16)0.0f;
        }
        *(v4bf*)&sAct[swz(r, c4)] = o;
    }
    // ---- stage Wa transposed (coalesced float4 reads): sW[n][k] = Wa[k][n] ----
    for (int idx4 = tid; idx4 < DIM * DIM / 4; idx4 += 256) {
        int k  = idx4 >> 5;
        int n4 = (idx4 & 31) * 4;
        const float4 wv = *(const float4*)&Wa[k * DIM + n4];
        sW[swz(n4 + 0, k)] = (__bf16)wv.x;
        sW[swz(n4 + 1, k)] = (__bf16)wv.y;
        sW[swz(n4 + 2, k)] = (__bf16)wv.z;
        sW[swz(n4 + 3, k)] = (__bf16)wv.w;
    }
    __syncthreads();

    const int wave = tid >> 5;
    const int lane = tid & 31;
    const int lm   = lane & 15;     // row/col within 16
    const int half = lane >> 4;     // 0: low K half, 1: high K half
    const int m0   = wave * 16;     // this wave's row strip

    // ---- build stage-1 A fragments (16x32 bf16 per k-chunk) ----
    v16bf a1[4];
    {
        const int m = m0 + lm;
        #pragma unroll
        for (int c = 0; c < 4; ++c) {
            v8bf lo = *(const v8bf*)&sAct[swz(m, c * 32 + half * 16)];
            v8bf hi = *(const v8bf*)&sAct[swz(m, c * 32 + half * 16 + 8)];
            a1[c] = __builtin_shufflevector(lo, hi,
                0,1,2,3,4,5,6,7,8,9,10,11,12,13,14,15);
        }
    }
    __syncthreads();   // frags in regs; sAct may now be overwritten

    // ---- stage 1: D = relu(A@Wa + ba) -> sAct ----
    #pragma unroll
    for (int nt = 0; nt < 8; ++nt) {
        const int n = nt * 16 + lm;
        const float bias = ba[n];
        v8f acc;
        #pragma unroll
        for (int v = 0; v < 8; ++v) acc[v] = bias;
        v16bf b[4];
        #pragma unroll
        for (int kt = 0; kt < 4; ++kt)
            b[kt] = *(const v16bf*)&sW[swz(n, kt * 32 + half * 16)];
        #pragma unroll
        for (int kt = 0; kt < 4; ++kt)
            acc = __builtin_amdgcn_wmma_f32_16x16x32_bf16(
                false, a1[kt], false, b[kt], (short)0, acc, false, false);
        #pragma unroll
        for (int v = 0; v < 8; ++v) {
            int m = m0 + v + half * 8;
            float r = acc[v];
            sAct[swz(m, n)] = (__bf16)(r > 0.0f ? r : 0.0f);
        }
    }
    __syncthreads();

    // ---- re-stage sW with Wb; build stage-2 A fragments ----
    for (int idx4 = tid; idx4 < DIM * DIM / 4; idx4 += 256) {
        int k  = idx4 >> 5;
        int n4 = (idx4 & 31) * 4;
        const float4 wv = *(const float4*)&Wb[k * DIM + n4];
        sW[swz(n4 + 0, k)] = (__bf16)wv.x;
        sW[swz(n4 + 1, k)] = (__bf16)wv.y;
        sW[swz(n4 + 2, k)] = (__bf16)wv.z;
        sW[swz(n4 + 3, k)] = (__bf16)wv.w;
    }
    v16bf a2[4];
    {
        const int m = m0 + lm;
        #pragma unroll
        for (int c = 0; c < 4; ++c) {
            v8bf lo = *(const v8bf*)&sAct[swz(m, c * 32 + half * 16)];
            v8bf hi = *(const v8bf*)&sAct[swz(m, c * 32 + half * 16 + 8)];
            a2[c] = __builtin_shufflevector(lo, hi,
                0,1,2,3,4,5,6,7,8,9,10,11,12,13,14,15);
        }
    }
    __syncthreads();

    // ---- stage 2: out = relu(H@Wb + bb) -> global ----
    #pragma unroll
    for (int nt = 0; nt < 8; ++nt) {
        const int n = nt * 16 + lm;
        const float bias = bb[n];
        v8f acc;
        #pragma unroll
        for (int v = 0; v < 8; ++v) acc[v] = bias;
        v16bf b[4];
        #pragma unroll
        for (int kt = 0; kt < 4; ++kt)
            b[kt] = *(const v16bf*)&sW[swz(n, kt * 32 + half * 16)];
        #pragma unroll
        for (int kt = 0; kt < 4; ++kt)
            acc = __builtin_amdgcn_wmma_f32_16x16x32_bf16(
                false, a2[kt], false, b[kt], (short)0, acc, false, false);
        #pragma unroll
        for (int v = 0; v < 8; ++v) {
            int row = r0 + m0 + v + half * 8;
            float r = acc[v];
            if (row < nRows)
                out[(long long)row * DIM + n] = (r > 0.0f ? r : 0.0f);
        }
    }
}

// ---------------------------------------------------------------------------
// pooling: pooled[batch[i]] += h[i]; counts[batch[i]] += 1
// ---------------------------------------------------------------------------
__global__ __launch_bounds__(256) void gin_pool(
    const float* __restrict__ h, const long long* __restrict__ batch,
    float* __restrict__ pooled, float* __restrict__ counts, int nRows)
{
    int i = __builtin_amdgcn_readfirstlane(blockIdx.x * 8 + (threadIdx.x >> 5));
    int j = threadIdx.x & 31;
    if (i >= nRows) return;
    int g = (int)batch[i];   // wave-uniform -> scalar load
    const float4 v = *(const float4*)&h[(long long)i * DIM + j * 4];
    float* base = &pooled[g * DIM + j * 4];
    unsafeAtomicAdd(base + 0, v.x);
    unsafeAtomicAdd(base + 1, v.y);
    unsafeAtomicAdd(base + 2, v.z);
    unsafeAtomicAdd(base + 3, v.w);
    if (j == 0) unsafeAtomicAdd(&counts[g], 1.0f);
}

// ---------------------------------------------------------------------------
// fc + sigmoid: out[g,c] = sigmoid(mean_g @ Wfc + bfc)
// ---------------------------------------------------------------------------
__global__ void gin_fc(const float* __restrict__ pooled,
                       const float* __restrict__ counts,
                       const float* __restrict__ Wfc,
                       const float* __restrict__ bfc,
                       float* __restrict__ out)
{
    int t = blockIdx.x * blockDim.x + threadIdx.x;
    if (t >= N_GRAPHS_C * N_CLASSES_C) return;
    int g = t / N_CLASSES_C;
    int c = t % N_CLASSES_C;
    float inv = 1.0f / fmaxf(counts[g], 1.0f);
    float z = bfc[c];
    for (int k = 0; k < DIM; ++k)
        z += pooled[g * DIM + k] * inv * Wfc[k * N_CLASSES_C + c];
    out[t] = 1.0f / (1.0f + __expf(-z));
}

// ---------------------------------------------------------------------------
extern "C" void kernel_launch(void* const* d_in, const int* in_sizes, int n_in,
                              void* d_out, int out_size, void* d_ws, size_t ws_size,
                              hipStream_t stream)
{
    const float*     x    = (const float*)d_in[0];
    const long long* eidx = (const long long*)d_in[1];
    const long long* batch= (const long long*)d_in[2];
    const float* W1a = (const float*)d_in[3];  const float* b1a = (const float*)d_in[4];
    const float* W1b = (const float*)d_in[5];  const float* b1b = (const float*)d_in[6];
    const float* W2a = (const float*)d_in[7];  const float* b2a = (const float*)d_in[8];
    const float* W2b = (const float*)d_in[9];  const float* b2b = (const float*)d_in[10];
    const float* Wfc = (const float*)d_in[11]; const float* bfc = (const float*)d_in[12];
    float* out = (float*)d_out;

    const long long nodeElems = (long long)N_NODES_C * DIM;   // 6.4M
    float* agg    = (float*)d_ws;
    float* h1     = agg + nodeElems;
    float* h2     = h1 + nodeElems;
    float* pooled = h2 + nodeElems;                           // 64*128
    float* counts = pooled + (long long)N_GRAPHS_C * DIM;     // 64

    const long long* src = eidx;
    const long long* dst = eidx + N_EDGES_C;

    const int zeroBlocks   = (int)((nodeElems + 255) / 256);
    const int scatBlocks   = (N_EDGES_C + 7) / 8;             // 8 edges / block
    const int mlpBlocks    = (N_NODES_C + DIM - 1) / DIM;     // 391
    const int poolBlocks   = (N_NODES_C + 7) / 8;             // 8 nodes / block
    const int poolZeroN    = N_GRAPHS_C * DIM + N_GRAPHS_C;

    // ---- layer 1 ----
    gin_zero_f32<<<zeroBlocks, 256, 0, stream>>>(agg, (int)nodeElems);
    gin_scatter<<<scatBlocks, 256, 0, stream>>>(x, src, dst, agg, N_EDGES_C);
    gin_mlp<<<mlpBlocks, 256, 0, stream>>>(x, agg, W1a, b1a, W1b, b1b, h1, N_NODES_C);

    // ---- layer 2 ----
    gin_zero_f32<<<zeroBlocks, 256, 0, stream>>>(agg, (int)nodeElems);
    gin_scatter<<<scatBlocks, 256, 0, stream>>>(h1, src, dst, agg, N_EDGES_C);
    gin_mlp<<<mlpBlocks, 256, 0, stream>>>(h1, agg, W2a, b2a, W2b, b2b, h2, N_NODES_C);

    // ---- pool + fc ----
    gin_zero_f32<<<(poolZeroN + 255) / 256, 256, 0, stream>>>(pooled, poolZeroN);
    gin_pool<<<poolBlocks, 256, 0, stream>>>(h2, batch, pooled, counts, N_NODES_C);
    gin_fc<<<(N_GRAPHS_C * N_CLASSES_C + 63) / 64, 64, 0, stream>>>(pooled, counts, Wfc, bfc, out);
}
